// BiLSTM_CRF_47064251629909
// MI455X (gfx1250) — compile-verified, matrix operand order checked by reading, not run on hardware
//
#include <hip/hip_runtime.h>
#include <hip/hip_bf16.h>

typedef __attribute__((ext_vector_type(16))) _Float16 v16h;
typedef __attribute__((ext_vector_type(8)))  float    v8f;

#define BB   128
#define SS   512
#define EE   256
#define HH   256
#define HD   512
#define TT   32
#define FOURH 1024
#define START_TAG 30
#define END_TAG   31

// ---------------------------------------------------------------------------
// WMMA fragment load helpers (16-bit A/B layout, wave32).
// A 16x32 f16 tile: lane l holds row m = l%16; element i holds
// k = (i/8)*16 + (l/16)*8 + i%8  -> two contiguous 8-half (16B) runs.
// B (from row-major W[N,K], used as K x N) has the identical per-lane layout
// with n = l%16 and k contiguous along W's rows.
// ---------------------------------------------------------------------------
__device__ __forceinline__ v16h load_frag(const _Float16* __restrict__ base, int ld, int lane) {
    int m  = lane & 15;
    int kh = (lane >> 4) << 3;
    const _Float16* p = base + (size_t)m * (size_t)ld + kh;
    union { v16h v; int4 q[2]; } u;
    u.q[0] = *(const int4*)(p);
    u.q[1] = *(const int4*)(p + 16);
    return u.v;
}

__device__ __forceinline__ float sigmf(float x) { return 1.f / (1.f + __expf(-x)); }

// ---------------------------------------------------------------------------
// f32 -> f16 conversion
// ---------------------------------------------------------------------------
__global__ void k_f32_to_f16(const float* __restrict__ src, _Float16* __restrict__ dst, int n) {
    int i = blockIdx.x * 256 + threadIdx.x;
    if (i < n) dst[i] = (_Float16)src[i];
}

// ---------------------------------------------------------------------------
// Embedding gather: x[B,S] int32 -> emb_f16 [S,B,E]
// ---------------------------------------------------------------------------
__global__ void k_gather_emb(const int* __restrict__ x, const float* __restrict__ emb,
                             _Float16* __restrict__ out) {
    size_t i = (size_t)blockIdx.x * 256 + threadIdx.x;   // over S*B*E
    int e  = (int)(i % EE);
    size_t sb = i / EE;
    int b  = (int)(sb % BB);
    int s  = (int)(sb / BB);
    int tok = x[(size_t)b * SS + s];
    out[i] = (_Float16)emb[(size_t)tok * EE + e];
}

// ---------------------------------------------------------------------------
// Input-projection GEMM (both directions): C[M,4H] = A[M,E] * W[4H,E]^T + bias
// A = emb_f16 [S*B, E]; wave computes a 16x64 tile; block = 4 waves (16x256).
// grid: (M/16, 4H/256, 2)
// ---------------------------------------------------------------------------
__global__ void k_gemm_preact(const _Float16* __restrict__ A,
                              const _Float16* __restrict__ W0, const _Float16* __restrict__ W1,
                              const float* __restrict__ bih0, const float* __restrict__ bhh0,
                              const float* __restrict__ bih1, const float* __restrict__ bhh1,
                              _Float16* __restrict__ C0, _Float16* __restrict__ C1) {
    const int K = EE, N = FOURH;
    int dir  = blockIdx.z;
    const _Float16* W = dir ? W1 : W0;
    const float* bih  = dir ? bih1 : bih0;
    const float* bhh  = dir ? bhh1 : bhh0;
    _Float16* C       = dir ? C1 : C0;

    int wave = threadIdx.x >> 5;
    int lane = threadIdx.x & 31;
    size_t m0 = (size_t)blockIdx.x * 16;
    int n0 = blockIdx.y * 256 + wave * 64;

    v8f acc[4] = {v8f{}, v8f{}, v8f{}, v8f{}};
    for (int kc = 0; kc < K; kc += 32) {
        v16h a = load_frag(A + m0 * K + kc, K, lane);
#pragma unroll
        for (int j = 0; j < 4; ++j) {
            v16h b = load_frag(W + (size_t)(n0 + j * 16) * K + kc, K, lane);
            acc[j] = __builtin_amdgcn_wmma_f32_16x16x32_f16(false, a, false, b,
                                                            (short)0, acc[j], false, false);
        }
    }
    int rbase = (lane >> 4) << 3;
    int cn = lane & 15;
#pragma unroll
    for (int j = 0; j < 4; ++j)
#pragma unroll
        for (int r = 0; r < 8; ++r) {
            size_t row = m0 + rbase + r;
            int col = n0 + j * 16 + cn;
            C[row * N + col] = (_Float16)(acc[j][r] + bih[col] + bhh[col]);
        }
}

// ---------------------------------------------------------------------------
// One LSTM time step, both directions (grid.z). Block = 128 threads = 4 waves;
// wave g owns gate g. Tile: 16 batch rows x 64 hidden cols.
// gates = preact[s] + h_prev @ Whh^T ; pointwise via LDS.
// grid: (B/16, H/64, 2)
// ---------------------------------------------------------------------------
__global__ void k_lstm_step(const _Float16* __restrict__ preact_f,
                            const _Float16* __restrict__ preact_b,
                            const _Float16* __restrict__ Whh_f,
                            const _Float16* __restrict__ Whh_b,
                            _Float16* __restrict__ h_all,   // [S,B,HD]
                            float* __restrict__ c_f, float* __restrict__ c_b,
                            int t) {
    __shared__ float lds_g[4][16][64];

    int dir = blockIdx.z;
    int s   = dir ? (SS - 1 - t) : t;
    int sp  = dir ? (s + 1) : (s - 1);
    const _Float16* preact = (dir ? preact_b : preact_f) + (size_t)s * BB * FOURH;
    const _Float16* Whh    = dir ? Whh_b : Whh_f;
    float* c               = dir ? c_b : c_f;

    int g    = threadIdx.x >> 5;     // wave = gate (i,f,g,o)
    int lane = threadIdx.x & 31;
    int m0   = blockIdx.x * 16;      // batch tile
    int nblk = blockIdx.y * 64;      // hidden-col tile

    v8f acc[4] = {v8f{}, v8f{}, v8f{}, v8f{}};
    if (t > 0) {
        const _Float16* Abase = h_all + ((size_t)sp * BB + m0) * HD + dir * HH;
#pragma unroll
        for (int kc = 0; kc < HH; kc += 32) {
            v16h a = load_frag(Abase + kc, HD, lane);
#pragma unroll
            for (int j = 0; j < 4; ++j) {
                v16h b = load_frag(Whh + (size_t)(g * HH + nblk + j * 16) * HH + kc, HH, lane);
                acc[j] = __builtin_amdgcn_wmma_f32_16x16x32_f16(false, a, false, b,
                                                                (short)0, acc[j], false, false);
            }
        }
    }
    int rbase = (lane >> 4) << 3;
    int cn = lane & 15;
#pragma unroll
    for (int j = 0; j < 4; ++j)
#pragma unroll
        for (int r = 0; r < 8; ++r) {
            int row = rbase + r;
            int col = j * 16 + cn;
            float pv = (float)preact[(size_t)(m0 + row) * FOURH + g * HH + nblk + col];
            lds_g[g][row][col] = acc[j][r] + pv;
        }
    __syncthreads();

    for (int e = threadIdx.x; e < 16 * 64; e += 128) {
        int m = e >> 6, nl = e & 63;
        float ig = sigmf(lds_g[0][m][nl]);
        float fg = sigmf(lds_g[1][m][nl]);
        float gg = tanhf(lds_g[2][m][nl]);
        float og = sigmf(lds_g[3][m][nl]);
        int b = m0 + m;
        int hc = nblk + nl;
        float cold = c[(size_t)b * HH + hc];       // memset to 0 each launch
        float cnew = fg * cold + ig * gg;
        c[(size_t)b * HH + hc] = cnew;
        float h = og * tanhf(cnew);
        h_all[((size_t)s * BB + b) * HD + dir * HH + hc] = (_Float16)h;
    }
}

// ---------------------------------------------------------------------------
// Emissions GEMM: C[M,32] = h_all[M,512] * W_tag[32,512]^T + b_tag, f32 out.
// Wave computes one 16x16 tile; block = 4 waves. grid: (M/16 * 2 / 4)
// ---------------------------------------------------------------------------
__global__ void k_gemm_emit(const _Float16* __restrict__ A, const _Float16* __restrict__ W,
                            const float* __restrict__ bias, float* __restrict__ C) {
    const int K = HD;
    int wave = threadIdx.x >> 5;
    int lane = threadIdx.x & 31;
    int flat = blockIdx.x * 4 + wave;
    size_t m0 = (size_t)(flat >> 1) * 16;
    int n0 = (flat & 1) * 16;

    v8f acc = v8f{};
#pragma unroll
    for (int kc = 0; kc < K; kc += 32) {
        v16h a = load_frag(A + m0 * K + kc, K, lane);
        v16h b = load_frag(W + (size_t)n0 * K + kc, K, lane);
        acc = __builtin_amdgcn_wmma_f32_16x16x32_f16(false, a, false, b,
                                                     (short)0, acc, false, false);
    }
    int rbase = (lane >> 4) << 3;
    int cn = lane & 15;
#pragma unroll
    for (int r = 0; r < 8; ++r) {
        size_t row = m0 + rbase + r;
        int col = n0 + cn;
        C[row * TT + col] = acc[r] + bias[col];
    }
}

// ---------------------------------------------------------------------------
// CRF forward logZ: one wave (32 lanes = 32 tags) per batch row.
// emissions [S,B,T]; streaming logsumexp over the 512-step scan.
// ---------------------------------------------------------------------------
__global__ void k_crf_fwd(const float* __restrict__ emissions, const float* __restrict__ trans,
                          float* __restrict__ fwd) {
    __shared__ float tr[TT][TT];
    int lane = threadIdx.x;
    int b = blockIdx.x;
    for (int i = 0; i < TT; ++i) tr[i][lane] = trans[i * TT + lane];
    __syncthreads();

    float alpha = (lane == START_TAG) ? 0.f : -10000.f;
    for (int s = 0; s < SS; ++s) {
        float emit = emissions[((size_t)s * BB + b) * TT + lane];
        float m = -3.0e38f;
#pragma unroll 4
        for (int i = 0; i < TT; ++i) {
            float v = __shfl(alpha, i, 32) + tr[i][lane];
            m = fmaxf(m, v);
        }
        float sum = 0.f;
#pragma unroll 4
        for (int i = 0; i < TT; ++i) {
            float v = __shfl(alpha, i, 32) + tr[i][lane];
            sum += __expf(v - m);
        }
        alpha = m + __logf(sum) + emit;
    }
    alpha += tr[END_TAG][lane];
    float m = alpha;
    for (int o = 16; o; o >>= 1) m = fmaxf(m, __shfl_xor(m, o, 32));
    float s = __expf(alpha - m);
    for (int o = 16; o; o >>= 1) s += __shfl_xor(s, o, 32);
    if (lane == 0) fwd[b] = m + __logf(s);
}

// ---------------------------------------------------------------------------
// Gold path score: one wave per batch row, lanes stride over S.
// ---------------------------------------------------------------------------
__global__ void k_gold(const float* __restrict__ emissions, const int* __restrict__ tags,
                       const float* __restrict__ trans, float* __restrict__ gold) {
    int lane = threadIdx.x;
    int b = blockIdx.x;
    float acc = 0.f;
    for (int s = lane; s < SS; s += 32) {
        int tag  = tags[(size_t)b * SS + s];
        int prev = (s == 0) ? START_TAG : tags[(size_t)b * SS + s - 1];
        acc += emissions[((size_t)s * BB + b) * TT + tag] + trans[tag * TT + prev];
    }
    for (int o = 16; o; o >>= 1) acc += __shfl_xor(acc, o, 32);
    if (lane == 0)
        gold[b] = acc + trans[END_TAG * TT + tags[(size_t)b * SS + SS - 1]];
}

__global__ void k_finalize(const float* __restrict__ fwd, const float* __restrict__ gold,
                           float* __restrict__ out) {
    __shared__ float buf[128];
    int t = threadIdx.x;
    buf[t] = fwd[t] - gold[t];
    __syncthreads();
    for (int s = 64; s; s >>= 1) {
        if (t < s) buf[t] += buf[t + s];
        __syncthreads();
    }
    if (t == 0) out[0] = buf[0] / (float)BB;
}

// ---------------------------------------------------------------------------
extern "C" void kernel_launch(void* const* d_in, const int* in_sizes, int n_in,
                              void* d_out, int out_size, void* d_ws, size_t ws_size,
                              hipStream_t stream) {
    (void)in_sizes; (void)n_in; (void)out_size; (void)ws_size;
    const int*   x     = (const int*)  d_in[0];
    const int*   tags  = (const int*)  d_in[1];
    const float* emb   = (const float*)d_in[2];
    const float* Wih_f = (const float*)d_in[3];
    const float* Whh_f = (const float*)d_in[4];
    const float* bih_f = (const float*)d_in[5];
    const float* bhh_f = (const float*)d_in[6];
    const float* Wih_b = (const float*)d_in[7];
    const float* Whh_b = (const float*)d_in[8];
    const float* bih_b = (const float*)d_in[9];
    const float* bhh_b = (const float*)d_in[10];
    const float* W_tag = (const float*)d_in[11];
    const float* b_tag = (const float*)d_in[12];
    const float* trans = (const float*)d_in[13];
    float* out = (float*)d_out;

    char* w = (char*)d_ws;
    size_t off = 0;
    auto alloc = [&](size_t bytes) -> void* {
        void* p = w + off;
        off += (bytes + 255) & ~(size_t)255;
        return p;
    };
    _Float16* emb16    = (_Float16*)alloc((size_t)SS * BB * EE * 2);
    _Float16* preactF  = (_Float16*)alloc((size_t)SS * BB * FOURH * 2);
    _Float16* preactB  = (_Float16*)alloc((size_t)SS * BB * FOURH * 2);
    _Float16* h_all    = (_Float16*)alloc((size_t)SS * BB * HD * 2);
    float*    emit     = (float*)   alloc((size_t)SS * BB * TT * 4);
    float*    cF       = (float*)   alloc((size_t)BB * HH * 4);
    float*    cB       = (float*)   alloc((size_t)BB * HH * 4);
    _Float16* wihF16f  = (_Float16*)alloc((size_t)FOURH * EE * 2);
    _Float16* wihF16b  = (_Float16*)alloc((size_t)FOURH * EE * 2);
    _Float16* whhF16f  = (_Float16*)alloc((size_t)FOURH * HH * 2);
    _Float16* whhF16b  = (_Float16*)alloc((size_t)FOURH * HH * 2);
    _Float16* wtag16   = (_Float16*)alloc((size_t)TT * HD * 2);
    float*    fwd      = (float*)   alloc((size_t)BB * 4);
    float*    gold     = (float*)   alloc((size_t)BB * 4);

    // Weight conversions f32 -> f16
    k_f32_to_f16<<<(FOURH * EE + 255) / 256, 256, 0, stream>>>(Wih_f, wihF16f, FOURH * EE);
    k_f32_to_f16<<<(FOURH * EE + 255) / 256, 256, 0, stream>>>(Wih_b, wihF16b, FOURH * EE);
    k_f32_to_f16<<<(FOURH * HH + 255) / 256, 256, 0, stream>>>(Whh_f, whhF16f, FOURH * HH);
    k_f32_to_f16<<<(FOURH * HH + 255) / 256, 256, 0, stream>>>(Whh_b, whhF16b, FOURH * HH);
    k_f32_to_f16<<<(TT * HD + 255) / 256, 256, 0, stream>>>(W_tag, wtag16, TT * HD);

    // Embedding gather (time-major, f16)
    k_gather_emb<<<(SS * BB * EE) / 256, 256, 0, stream>>>(x, emb, emb16);

    // Big parallel input-projection GEMM for both directions
    {
        dim3 grid((SS * BB) / 16, FOURH / 256, 2);
        k_gemm_preact<<<grid, 128, 0, stream>>>(emb16, wihF16f, wihF16b,
                                                bih_f, bhh_f, bih_b, bhh_b,
                                                preactF, preactB);
    }

    // Zero cell states, then run the sequential recurrence (fwd+bwd fused)
    hipMemsetAsync(cF, 0, (size_t)BB * HH * 4, stream);
    hipMemsetAsync(cB, 0, (size_t)BB * HH * 4, stream);
    {
        dim3 grid(BB / 16, HH / 64, 2);
        for (int t = 0; t < SS; ++t)
            k_lstm_step<<<grid, 128, 0, stream>>>(preactF, preactB, whhF16f, whhF16b,
                                                  h_all, cF, cB, t);
    }

    // Emissions GEMM: [S*B, 512] x [32, 512]^T
    k_gemm_emit<<<((SS * BB) / 16 * 2) / 4, 128, 0, stream>>>(h_all, wtag16, b_tag, emit);

    // CRF forward scan + gold score + mean
    k_crf_fwd<<<BB, 32, 0, stream>>>(emit, trans, fwd);
    k_gold<<<BB, 32, 0, stream>>>(emit, tags, trans, gold);
    k_finalize<<<1, 128, 0, stream>>>(fwd, gold, out);
}
